// Core_Datadriven_47021301956913
// MI455X (gfx1250) — matrix operand order repaired and checked by reading, not compiled, not run on hardware
//
#include <hip/hip_runtime.h>
#include <math.h>

typedef __attribute__((ext_vector_type(2)))  float  v2f;
typedef __attribute__((ext_vector_type(8)))  float  v8f;
typedef __attribute__((ext_vector_type(16))) __bf16 v16bf;

#define B_    16
#define DIN   128
#define T_    12
#define N_    512
#define TN    6144            // T_*N_
#define PERB  (DIN*TN)        // 786432
#define DCORE 128
#define CORES 32
#define DCAT  256
#define DHID  1024
#define DOUT  128

static __device__ __forceinline__ v8f wmma4(v2f a, v2f b, v8f c) {
  // D = A(16x4) * B(4x16) + C(16x16), fp32
  return __builtin_amdgcn_wmma_f32_16x16x4_f32(false, a, false, b, (short)0, c,
                                               false, false);
}
static __device__ __forceinline__ v8f wmma32(v16bf a, v16bf b, v8f c) {
  // D = A(16x32 bf16) * B(32x16 bf16) + C(16x16 f32)
  return __builtin_amdgcn_wmma_f32_16x16x32_bf16(false, a, false, b, (short)0, c,
                                                 false, false);
}

// ---------------------------------------------------------------------------
// Pack Wv[128,128] into f32-WMMA-A lane order:
// P[((mb*(K/4)+ks)*32 + L)*2 + j] = W[mb*16 + (L&15)][ks*4 + (L>>4)*2 + j]
// ---------------------------------------------------------------------------
__global__ void kp_pack(const float* __restrict__ Wv, float* __restrict__ pWv) {
  int idx = blockIdx.x * 256 + threadIdx.x;
  if (idx >= 16384) return;
  const int K = 128, kq = K >> 2;
  int j  = idx & 1;
  int L  = (idx >> 1) & 31;
  int rest = idx >> 6;
  int ks = rest % kq, mb = rest / kq;
  int m  = mb * 16 + (L & 15);
  int k  = ks * 4 + ((L >> 4) << 1) + j;
  pWv[idx] = Wv[m * K + k];
}

// ---------------------------------------------------------------------------
// Pack W1[1024,256], W2[128,1024] into bf16 hi/lo WMMA-A lane order (k-step 32):
// per lane 16 halves: e<8 -> k = ks*32 + lh*8 + e ; e>=8 -> k = ks*32+16+lh*8+(e-8)
// ---------------------------------------------------------------------------
__global__ void kp_pack_bf16(const float* __restrict__ W1, const float* __restrict__ W2,
                             __bf16* __restrict__ pW1h, __bf16* __restrict__ pW1l,
                             __bf16* __restrict__ pW2h, __bf16* __restrict__ pW2l) {
  int gid = blockIdx.x * 256 + threadIdx.x;
  const float* W; __bf16 *Ph, *Pl; int K; int idx;
  if (gid < 262144)      { W = W1; Ph = pW1h; Pl = pW1l; K = 256;  idx = gid; }
  else if (gid < 393216) { W = W2; Ph = pW2h; Pl = pW2l; K = 1024; idx = gid - 262144; }
  else return;
  int e  = idx & 15;
  int L  = (idx >> 4) & 31;
  int rest = idx >> 9;
  int kq = K >> 5;
  int ks = rest % kq, mb = rest / kq;
  int m  = mb * 16 + (L & 15);
  int lh = L >> 4;
  int k  = ks * 32 + ((e < 8) ? (lh * 8 + e) : (16 + lh * 8 + (e - 8)));
  float wv = W[m * K + k];
  __bf16 h = (__bf16)wv;
  Ph[idx] = h;
  Pl[idx] = (__bf16)(wv - (float)h);
}

// ---------------------------------------------------------------------------
// xsum[b,i,n] = sum_t input[b,i,t,n]
// ---------------------------------------------------------------------------
__global__ void k1_xsum(const float* __restrict__ x, float* __restrict__ xsum) {
  int gid = blockIdx.x * 256 + threadIdx.x;       // 16*128*512 threads
  int n = gid & 511, i = (gid >> 9) & 127, b = gid >> 16;
  const float* p = x + (size_t)b * PERB + (size_t)i * TN + n;
  float acc = 0.f;
#pragma unroll
  for (int t = 0; t < T_; ++t) acc += p[t * N_];
  xsum[((b * DIN) + i) * N_ + n] = acc;
}

// ---------------------------------------------------------------------------
// qsum[b,f,n] = sum_i Wq[f,i]*xsum[b,i,n] + T*bq[f]
// ---------------------------------------------------------------------------
__global__ void k2_qsum(const float* __restrict__ xsum, const float* __restrict__ Wq,
                        const float* __restrict__ bq, float* __restrict__ qsum) {
  int gid = blockIdx.x * 256 + threadIdx.x;
  int n = gid & 511, f = (gid >> 9) & 127, b = gid >> 16;
  float acc = (float)T_ * bq[f];
  const float* wr = Wq + f * DIN;
  const float* xc = xsum + (size_t)b * DIN * N_ + n;
#pragma unroll 8
  for (int i = 0; i < DIN; ++i) acc += wr[i] * xc[i * N_];
  qsum[((b * DCORE) + f) * N_ + n] = acc;
}

// ---------------------------------------------------------------------------
// v = Wv @ x + bv  per batch: [128,128]@[128,6144].  WMMA f32 16x16x4.
// ---------------------------------------------------------------------------
__global__ void kv_gemm(const float* __restrict__ x, const float* __restrict__ pWv,
                        const float* __restrict__ bv, float* __restrict__ v) {
  __shared__ float xt[32][132];   // column-major tile: xt[col][k]
  int b = blockIdx.y;
  int col0 = blockIdx.x * 32;
  int tid = threadIdx.x;

  for (int idx = tid; idx < DIN * 32; idx += 256) {
    int i = idx >> 5, jj = idx & 31;
    xt[jj][i] = x[(size_t)b * PERB + (size_t)i * TN + col0 + jj];
  }
  __syncthreads();

  int w = tid >> 5, lane = tid & 31;
  int lr = lane & 15, lh = lane >> 4;
  int m0 = w * 16;

  v8f acc0, acc1;
#pragma unroll
  for (int r = 0; r < 8; ++r) {
    float bb = bv[m0 + lh * 8 + r];
    acc0[r] = bb; acc1[r] = bb;
  }

  const float* aBase = pWv + ((size_t)w * 32) * 64;
#pragma unroll 4
  for (int ks = 0; ks < 32; ++ks) {
    v2f a = *(const v2f*)(aBase + ((size_t)ks * 32 + lane) * 2);
    int row = ks * 4 + lh * 2;
    v2f b0 = *(const v2f*)(&xt[lr][row]);
    v2f b1 = *(const v2f*)(&xt[16 + lr][row]);
    acc0 = wmma4(a, b0, acc0);
    acc1 = wmma4(a, b1, acc1);
  }

#pragma unroll
  for (int r = 0; r < 8; ++r) {
    int f = m0 + lh * 8 + r;
    float* o = v + (size_t)b * PERB + (size_t)f * TN + col0;
    o[lr]      = acc0[r];
    o[16 + lr] = acc1[r];
  }
}

// ---------------------------------------------------------------------------
// Softmax mixing per (b,f):  aff[c,n] = cores[c,f]*qsum[b,f,n]/sqrt(128) (rank-1)
// ---------------------------------------------------------------------------
__global__ void k3_mix(const float* __restrict__ qsum, const float* __restrict__ cores,
                       const float* __restrict__ vbuf, float* __restrict__ vbout) {
  __shared__ float qs[512];
  __shared__ float vS[T_][512];
  __shared__ float cf[32];
  __shared__ float vagg[T_][32];
  __shared__ float red[256];
  __shared__ float sc[4];          // qmax, qmin, cmax, cmin
  __shared__ float psum[32][8];
  __shared__ float pv[T_][32][8];

  int f = blockIdx.x, b = blockIdx.y;
  int tid = threadIdx.x;
  const float s = 0.08838834764831845f;  // 1/sqrt(128)
  size_t qbase = ((size_t)b * DCORE + f) * N_;
  size_t vbase = ((size_t)b * DCORE + f) * (size_t)TN;

  qs[tid]       = qsum[qbase + tid];
  qs[tid + 256] = qsum[qbase + tid + 256];
  for (int idx = tid; idx < T_ * 512; idx += 256) {
    int t = idx >> 9, n = idx & 511;
    vS[t][n] = vbuf[vbase + (size_t)t * N_ + n];
  }
  if (tid < 32) cf[tid] = cores[tid * DCORE + f];
  __syncthreads();

  red[tid] = fmaxf(qs[tid], qs[tid + 256]); __syncthreads();
  for (int off = 128; off > 0; off >>= 1) {
    if (tid < off) red[tid] = fmaxf(red[tid], red[tid + off]);
    __syncthreads();
  }
  if (tid == 0) sc[0] = red[0];
  __syncthreads();
  red[tid] = fminf(qs[tid], qs[tid + 256]); __syncthreads();
  for (int off = 128; off > 0; off >>= 1) {
    if (tid < off) red[tid] = fminf(red[tid], red[tid + off]);
    __syncthreads();
  }
  if (tid == 0) {
    sc[1] = red[0];
    float cm = -1e30f, cn = 1e30f;
    for (int c = 0; c < 32; ++c) { cm = fmaxf(cm, cf[c]); cn = fminf(cn, cf[c]); }
    sc[2] = cm; sc[3] = cn;
  }
  __syncthreads();

  // Phase A: per core c, softmax over n; aggregate v -> vagg[t][c]
  {
    int c = tid >> 3, sub = tid & 7;
    float cc = cf[c] * s;
    float Mc = cc * ((cc >= 0.f) ? sc[0] : sc[1]);
    float ps = 0.f;
    float pva[T_];
#pragma unroll
    for (int t = 0; t < T_; ++t) pva[t] = 0.f;
    for (int nn = 0; nn < 64; ++nn) {
      int n = nn * 8 + sub;
      float e = __expf(cc * qs[n] - Mc);
      ps += e;
#pragma unroll
      for (int t = 0; t < T_; ++t) pva[t] += vS[t][n] * e;
    }
    psum[c][sub] = ps;
#pragma unroll
    for (int t = 0; t < T_; ++t) pv[t][c][sub] = pva[t];
  }
  __syncthreads();
  if (tid < 32) {
    int c = tid;
    float rs = 0.f;
    for (int k = 0; k < 8; ++k) rs += psum[c][k];
    float inv = 1.f / rs;
    for (int t = 0; t < T_; ++t) {
      float a = 0.f;
      for (int k = 0; k < 8; ++k) a += pv[t][c][k];
      vagg[t][c] = a * inv;
    }
  }
  __syncthreads();

  // Phase B: per node n, softmax over c; broadcast back
  for (int n = tid; n < 512; n += 256) {
    float q = qs[n] * s;
    float colmax = q * ((q >= 0.f) ? sc[2] : sc[3]);
    float e[32];
    float cs = 0.f;
#pragma unroll
    for (int c = 0; c < 32; ++c) { e[c] = __expf(cf[c] * q - colmax); cs += e[c]; }
    float inv = 1.f / cs;
#pragma unroll
    for (int t = 0; t < T_; ++t) {
      float acc = 0.f;
#pragma unroll
      for (int c = 0; c < 32; ++c) acc += vagg[t][c] * e[c];
      vbout[vbase + (size_t)t * N_ + n] = acc * inv;
    }
  }
}

// ---------------------------------------------------------------------------
// Fused MLP with split-bf16 (hi/lo) WMMA:  cat = [x - vb ; vb] (256 rows),
// h = gelu(W1@cat + b1) per 128-row block in LDS, out = W2@h + b2 in registers.
// Block: 256 threads, 64-column tile.  A*B = Ah*Bh + Ah*Bl + Al*Bh (fp32 acc).
// ---------------------------------------------------------------------------
__global__ void k4_mlp(const float* __restrict__ x, const float* __restrict__ vb,
                       const __bf16* __restrict__ pW1h, const __bf16* __restrict__ pW1l,
                       const float* __restrict__ b1,
                       const __bf16* __restrict__ pW2h, const __bf16* __restrict__ pW2l,
                       const float* __restrict__ b2, float* __restrict__ out) {
  __shared__ __bf16 catH[64][264];  // cat_t[col][k] hi, k in [0,256)
  __shared__ __bf16 catL[64][264];  // cat_t lo
  __shared__ __bf16 htH[64][136];   // h-block transposed hi, kr in [0,128)
  __shared__ __bf16 htL[64][136];   // h-block lo

  int b = blockIdx.y;
  int col0 = blockIdx.x * 64;
  int tid = threadIdx.x;
  int w = tid >> 5, lane = tid & 31;
  int lr = lane & 15, lh = lane >> 4;

  for (int idx = tid; idx < DIN * 64; idx += 256) {
    int i = idx >> 6, jj = idx & 63;
    size_t g = (size_t)b * PERB + (size_t)i * TN + col0 + jj;
    float xv = x[g], vv = vb[g];
    float c0 = xv - vv;
    __bf16 h0 = (__bf16)c0, h1 = (__bf16)vv;
    catH[jj][i]       = h0;  catL[jj][i]       = (__bf16)(c0 - (float)h0);
    catH[jj][DIN + i] = h1;  catL[jj][DIN + i] = (__bf16)(vv - (float)h1);
  }

  v8f oacc[4];
#pragma unroll
  for (int ct = 0; ct < 4; ++ct)
#pragma unroll
    for (int r = 0; r < 8; ++r) oacc[ct][r] = b2[w * 16 + lh * 8 + r];
  __syncthreads();

  for (int hb = 0; hb < 8; ++hb) {
    // ---- phase 2: 128 h-rows [hb*128, ...), this wave does 16 of them
    int mb = hb * 8 + w;
    v8f hacc[4];
#pragma unroll
    for (int ct = 0; ct < 4; ++ct)
#pragma unroll
      for (int r = 0; r < 8; ++r) hacc[ct][r] = b1[mb * 16 + lh * 8 + r];

#pragma unroll 2
    for (int ks = 0; ks < 8; ++ks) {          // K = 256, step 32
      size_t ao = ((size_t)(mb * 8 + ks) * 32 + lane) * 16;
      v16bf ah = *(const v16bf*)(pW1h + ao);
      v16bf al = *(const v16bf*)(pW1l + ao);
      int k0 = ks * 32 + lh * 8;
#pragma unroll
      for (int ct = 0; ct < 4; ++ct) {
        int c = ct * 16 + lr;
        typedef __attribute__((ext_vector_type(8))) __bf16 v8bf_;
        v8bf_ bh0 = *(const v8bf_*)(&catH[c][k0]), bh1 = *(const v8bf_*)(&catH[c][k0 + 16]);
        v8bf_ bl0 = *(const v8bf_*)(&catL[c][k0]), bl1 = *(const v8bf_*)(&catL[c][k0 + 16]);
        v16bf bh = __builtin_shufflevector(bh0, bh1, 0,1,2,3,4,5,6,7,8,9,10,11,12,13,14,15);
        v16bf bl = __builtin_shufflevector(bl0, bl1, 0,1,2,3,4,5,6,7,8,9,10,11,12,13,14,15);
        hacc[ct] = wmma32(ah, bh, hacc[ct]);
        hacc[ct] = wmma32(ah, bl, hacc[ct]);
        hacc[ct] = wmma32(al, bh, hacc[ct]);
      }
    }
    // exact gelu + hi/lo split to LDS (transposed)
#pragma unroll
    for (int ct = 0; ct < 4; ++ct) {
      int c = ct * 16 + lr;
      int kr = w * 16 + lh * 8;
#pragma unroll
      for (int r = 0; r < 8; ++r) {
        float hv = hacc[ct][r];
        float g = 0.5f * hv * (1.f + erff(hv * 0.70710678118654752f));
        __bf16 gh = (__bf16)g;
        htH[c][kr + r] = gh;
        htL[c][kr + r] = (__bf16)(g - (float)gh);
      }
    }
    __syncthreads();

    // ---- phase 3: out += W2[:, hb*128 : hb*128+128] @ hblock
#pragma unroll 2
    for (int ks = 0; ks < 4; ++ks) {          // 128 K, step 32
      size_t ao = ((size_t)(w * 32 + hb * 4 + ks) * 32 + lane) * 16;
      v16bf ah = *(const v16bf*)(pW2h + ao);
      v16bf al = *(const v16bf*)(pW2l + ao);
      int k0 = ks * 32 + lh * 8;
#pragma unroll
      for (int ct = 0; ct < 4; ++ct) {
        int c = ct * 16 + lr;
        typedef __attribute__((ext_vector_type(8))) __bf16 v8bf_;
        v8bf_ bh0 = *(const v8bf_*)(&htH[c][k0]), bh1 = *(const v8bf_*)(&htH[c][k0 + 16]);
        v8bf_ bl0 = *(const v8bf_*)(&htL[c][k0]), bl1 = *(const v8bf_*)(&htL[c][k0 + 16]);
        v16bf bh = __builtin_shufflevector(bh0, bh1, 0,1,2,3,4,5,6,7,8,9,10,11,12,13,14,15);
        v16bf bl = __builtin_shufflevector(bl0, bl1, 0,1,2,3,4,5,6,7,8,9,10,11,12,13,14,15);
        oacc[ct] = wmma32(ah, bh, oacc[ct]);
        oacc[ct] = wmma32(ah, bl, oacc[ct]);
        oacc[ct] = wmma32(al, bh, oacc[ct]);
      }
    }
    __syncthreads();
  }

#pragma unroll
  for (int ct = 0; ct < 4; ++ct)
#pragma unroll
    for (int r = 0; r < 8; ++r) {
      int o = w * 16 + lh * 8 + r;
      out[(size_t)b * PERB + (size_t)o * TN + col0 + ct * 16 + lr] = oacc[ct][r];
    }
}

// ---------------------------------------------------------------------------
extern "C" void kernel_launch(void* const* d_in, const int* in_sizes, int n_in,
                              void* d_out, int out_size, void* d_ws, size_t ws_size,
                              hipStream_t stream) {
  (void)in_sizes; (void)n_in; (void)out_size; (void)ws_size;
  const float* x     = (const float*)d_in[0];
  const float* Wq    = (const float*)d_in[1];
  const float* bq    = (const float*)d_in[2];
  const float* Wv    = (const float*)d_in[3];
  const float* bv    = (const float*)d_in[4];
  const float* cores = (const float*)d_in[5];
  const float* W1    = (const float*)d_in[6];
  const float* b1    = (const float*)d_in[7];
  const float* W2    = (const float*)d_in[8];
  const float* b2    = (const float*)d_in[9];
  float* out = (float*)d_out;

  float*  ws   = (float*)d_ws;
  float*  pWv  = ws;                          // 16384 f32
  __bf16* pW1h = (__bf16*)(pWv + 16384);      // 262144 bf16
  __bf16* pW1l = pW1h + 262144;
  __bf16* pW2h = pW1l + 262144;               // 131072 bf16
  __bf16* pW2l = pW2h + 131072;
  float*  xsum = (float*)(pW2l + 131072);     // 1048576 f32
  float*  qsum = xsum + 1048576;              // 1048576 f32
  float*  vbuf = qsum + 1048576;              // 12582912 f32
  float*  vbm  = vbuf + 12582912;             // 12582912 f32

  kp_pack      <<<64, 256, 0, stream>>>(Wv, pWv);
  kp_pack_bf16 <<<1536, 256, 0, stream>>>(W1, W2, pW1h, pW1l, pW2h, pW2l);
  k1_xsum      <<<4096, 256, 0, stream>>>(x, xsum);
  k2_qsum      <<<4096, 256, 0, stream>>>(xsum, Wq, bq, qsum);
  kv_gemm      <<<dim3(TN / 32, B_), 256, 0, stream>>>(x, pWv, bv, vbuf);
  k3_mix       <<<dim3(DCORE, B_), 256, 0, stream>>>(qsum, cores, vbuf, vbm);
  k4_mlp       <<<dim3(TN / 64, B_), 256, 0, stream>>>(x, vbm, pW1h, pW1l, b1,
                                                       pW2h, pW2l, b2, out);
}
